// DeepUnfolding_53094385713987
// MI455X (gfx1250) — matrix operand order repaired and checked by reading, not compiled, not run on hardware
//
#include <hip/hip_runtime.h>

typedef __attribute__((ext_vector_type(16))) _Float16 v16h;
typedef __attribute__((ext_vector_type(8)))  _Float16 v8h;
typedef __attribute__((ext_vector_type(8)))  float    v8f;

#define BATCH   32
#define NPULSE  128
#define LEN     4096
#define KTAPS   129
#define PADK    64
#define NSTEP   10
#define NTILES  (LEN/128)      // 32 column tiles of 128
#define LDSW    136            // padded LDS row stride in halves (17*8 -> keeps 16B align)

__device__ __forceinline__ v8f wmma_f16(v16h a, v16h b, v8f c) {
  // D = A(16x32 f16) * B(32x16 f16) + C(16x16 f32)
  return __builtin_amdgcn_wmma_f32_16x16x32_f16(false, a, false, b, (short)0, c, false, false);
}

__device__ __forceinline__ v16h concat8(v8h lo, v8h hi) {
  return __builtin_shufflevector(lo, hi, 0, 1, 2, 3, 4, 5, 6, 7, 8, 9, 10, 11, 12, 13, 14, 15);
}

// ---------------------------------------------------------------------------
// init: w_mag = 1, w_phase = w_phase_init
__global__ void init_w_kernel(const float* __restrict__ wp0,
                              float* __restrict__ wmag, float* __restrict__ wph) {
  int i = blockIdx.x * blockDim.x + threadIdx.x;
  if (i < BATCH * NPULSE) { wmag[i] = 1.0f; wph[i] = wp0[i]; }
}

// ---------------------------------------------------------------------------
// fftshift-folded DFT matrix: F[f,s] = exp(-2*pi*i*((f+64)%128)*s/128), f16
__global__ void dft_init_kernel(_Float16* __restrict__ Fr, _Float16* __restrict__ Fi) {
  int i = blockIdx.x * blockDim.x + threadIdx.x;
  if (i >= NPULSE * NPULSE) return;
  int f = i >> 7, s = i & 127;
  int fsh = (f + 64) & 127;
  int t = (fsh * s) & 127;                 // exact periodic reduction
  float ang = -6.28318530717958647692f * (float)t / 128.0f;
  float sn, cs; __sincosf(ang, &sn, &cs);
  Fr[i] = (_Float16)cs;
  Fi[i] = (_Float16)sn;
}

// ---------------------------------------------------------------------------
// matched filter: per-row complex 129-tap cross-correlation with conj(kernel)
// out_re = sum_k xr*kr + xi*ki ; out_im = sum_k xi*kr - xr*ki ; stored f16
__global__ __launch_bounds__(256)
void matched_filter_kernel(const float* __restrict__ xr, const float* __restrict__ xi,
                           const float* __restrict__ kr, const float* __restrict__ ki,
                           _Float16* __restrict__ outr, _Float16* __restrict__ outi) {
  __shared__ float sre[LEN + 2 * PADK];
  __shared__ float sim[LEN + 2 * PADK];
  __shared__ float skr[KTAPS];
  __shared__ float ski[KTAPS];
  const int row = blockIdx.x;                       // b*S + s
  const int tid = threadIdx.x;
  const float* xrr = xr + (size_t)row * LEN;
  const float* xir = xi + (size_t)row * LEN;
  for (int i = tid; i < LEN + 2 * PADK; i += 256) {
    int l = i - PADK;
    bool ok = (l >= 0) && (l < LEN);
    sre[i] = ok ? xrr[l] : 0.0f;
    sim[i] = ok ? xir[l] : 0.0f;
  }
  for (int i = tid; i < KTAPS; i += 256) { skr[i] = kr[i]; ski[i] = ki[i]; }
  __syncthreads();
  for (int l = tid; l < LEN; l += 256) {
    float ar = 0.0f, ai = 0.0f;
    #pragma unroll 4
    for (int k = 0; k < KTAPS; k++) {
      float xrv = sre[l + k], xiv = sim[l + k];
      float krv = skr[k], kiv = ski[k];
      ar = fmaf(xrv, krv, ar); ar = fmaf(xiv, kiv, ar);
      ai = fmaf(xiv, krv, ai); ai = fmaf(-xrv, kiv, ai);
    }
    outr[(size_t)row * LEN + l] = (_Float16)ar;
    outi[(size_t)row * LEN + l] = (_Float16)ai;
  }
}

// ---------------------------------------------------------------------------
// per-batch: softmax(w_mag) -> norm_mag, theta = atan(w_phase), c = nm*e^{-i th}
__global__ void compute_c_kernel(const float* __restrict__ wmag, const float* __restrict__ wph,
                                 float* __restrict__ cre, float* __restrict__ cim,
                                 float* __restrict__ pbuf) {
  __shared__ float red[NPULSE];
  int b = blockIdx.x, s = threadIdx.x;
  float wm = wmag[b * NPULSE + s];
  red[s] = wm; __syncthreads();
  #pragma unroll
  for (int st = 64; st > 0; st >>= 1) { if (s < st) red[s] = fmaxf(red[s], red[s + st]); __syncthreads(); }
  float mx = red[0]; __syncthreads();
  float ex = __expf(wm - mx);
  red[s] = ex; __syncthreads();
  #pragma unroll
  for (int st = 64; st > 0; st >>= 1) { if (s < st) red[s] += red[s + st]; __syncthreads(); }
  float p = ex / red[0];
  float nm = (float)NPULSE * p;
  float th = atanf(wph[b * NPULSE + s]);
  float sn, cs; __sincosf(th, &sn, &cs);
  cre[b * NPULSE + s] = nm * cs;
  cim[b * NPULSE + s] = -nm * sn;
  pbuf[b * NPULSE + s] = p;
}

// ---------------------------------------------------------------------------
// rd tile = F(128x128) * (c .* x_mf)(128x128cols) via WMMA; e = |rd|^2; argmax
// iq tile stored TRANSPOSED in LDS: siqT[l][s], so a lane's B fragment is
// 16 contiguous halves -> two aligned ds_load_b128 per fragment.
__global__ __launch_bounds__(256)
void dft_energy_kernel(const _Float16* __restrict__ xmr, const _Float16* __restrict__ xmi,
                       const _Float16* __restrict__ Fr, const _Float16* __restrict__ Fi,
                       const float* __restrict__ cre, const float* __restrict__ cim,
                       float* __restrict__ pmv, int* __restrict__ pmi) {
  __shared__ _Float16 sreT[NPULSE * LDSW];   // [l][s]
  __shared__ _Float16 simT[NPULSE * LDSW];
  __shared__ float scr[NPULSE];
  __shared__ float sci[NPULSE];
  __shared__ float redv[256];
  __shared__ int   redi[256];
  const int b = blockIdx.x / NTILES;
  const int tile = blockIdx.x % NTILES;
  const int tid = threadIdx.x;
  const int l0 = tile * 128;
  const size_t base = (size_t)b * NPULSE * LEN;

  if (tid < NPULSE) { scr[tid] = cre[b * NPULSE + tid]; sci[tid] = cim[b * NPULSE + tid]; }
  if (tid == 0) { __builtin_prefetch(Fr, 0, 0); __builtin_prefetch(Fi, 0, 0); }
  __syncthreads();

  // stage iq = c_s * x_mf tile, transposed, via 16-byte global loads
  for (int e = tid; e < NPULSE * 16; e += 256) {   // 128 rows x 16 chunks of 8 halves
    int s = e >> 4, c8 = e & 15;
    const v8h xr8 = *(const v8h*)(xmr + base + (size_t)s * LEN + l0 + c8 * 8);
    const v8h xi8 = *(const v8h*)(xmi + base + (size_t)s * LEN + l0 + c8 * 8);
    float cr = scr[s], ci = sci[s];
    #pragma unroll
    for (int q = 0; q < 8; q++) {
      int l = c8 * 8 + q;
      float xrv = (float)xr8[q], xiv = (float)xi8[q];
      sreT[l * LDSW + s] = (_Float16)(cr * xrv - ci * xiv);
      simT[l * LDSW + s] = (_Float16)(cr * xiv + ci * xrv);
    }
  }
  __syncthreads();

  const int wave = tid >> 5, lane = tid & 31;
  const int half = lane >> 4, m16 = lane & 15, n16 = lane & 15;
  const int f0 = wave * 16;

  // A fragments: 16 rows of F, 4 K-chunks, re & im.
  // Per lane/kc: halves [32kc+16h, +15] of row f0+m16 (16B aligned).
  v16h afr[4], afi[4];
  const _Float16* frow_r = Fr + (f0 + m16) * NPULSE;
  const _Float16* frow_i = Fi + (f0 + m16) * NPULSE;
  #pragma unroll
  for (int kc = 0; kc < 4; kc++) {
    int ka = 32 * kc + 16 * half;
    afr[kc] = concat8(*(const v8h*)(frow_r + ka), *(const v8h*)(frow_r + ka + 8));
    afi[kc] = concat8(*(const v8h*)(frow_i + ka), *(const v8h*)(frow_i + ka + 8));
  }

  float bmax = -1.0f; int bidx = 0x7fffffff;
  for (int nt = 0; nt < 8; nt++) {
    const int ncol = nt * 16 + n16;
    const _Float16* prow_r = &sreT[ncol * LDSW];
    const _Float16* prow_i = &simT[ncol * LDSW];
    v8f acc_rr = {};   // Fr*iqr
    v8f acc_ii = {};   // Fi*iqi   (subtract at end: f16 WMMA has no A/B negate)
    v8f acc_im = {};   // Fr*iqi + Fi*iqr
    #pragma unroll
    for (int kc = 0; kc < 4; kc++) {
      int kb = 32 * kc + 16 * half;   // 16 contiguous halves
      v16h br = concat8(*(const v8h*)(prow_r + kb), *(const v8h*)(prow_r + kb + 8));
      v16h bi = concat8(*(const v8h*)(prow_i + kb), *(const v8h*)(prow_i + kb + 8));
      acc_rr = wmma_f16(afr[kc], br, acc_rr);
      acc_ii = wmma_f16(afi[kc], bi, acc_ii);
      acc_im = wmma_f16(afr[kc], bi, acc_im);
      acc_im = wmma_f16(afi[kc], br, acc_im);
    }
    #pragma unroll
    for (int r = 0; r < 8; r++) {
      float re = acc_rr[r] - acc_ii[r];
      float im = acc_im[r];
      float ev = re * re + im * im;
      int f = f0 + r + 8 * half;
      int flat = f * LEN + l0 + nt * 16 + n16;
      if (ev > bmax || (ev == bmax && flat < bidx)) { bmax = ev; bidx = flat; }
    }
  }

  redv[tid] = bmax; redi[tid] = bidx; __syncthreads();
  #pragma unroll
  for (int st = 128; st > 0; st >>= 1) {
    if (tid < st) {
      float ov = redv[tid + st]; int oi = redi[tid + st];
      if (ov > redv[tid] || (ov == redv[tid] && oi < redi[tid])) { redv[tid] = ov; redi[tid] = oi; }
    }
    __syncthreads();
  }
  if (tid == 0) { pmv[blockIdx.x] = redv[0]; pmi[blockIdx.x] = redi[0]; }
}

// ---------------------------------------------------------------------------
// reduce per-tile maxima -> l* (argmax flat % L), one wave per batch
__global__ void reduce_lstar_kernel(const float* __restrict__ pmv, const int* __restrict__ pmi,
                                    int* __restrict__ lstar) {
  int b = blockIdx.x, t = threadIdx.x;         // 32 threads
  float v = pmv[b * NTILES + t];
  int   ix = pmi[b * NTILES + t];
  #pragma unroll
  for (int off = 16; off > 0; off >>= 1) {
    float ov = __shfl_down(v, off);
    int   oi = __shfl_down(ix, off);
    if (ov > v || (ov == v && oi < ix)) { v = ov; ix = oi; }
  }
  if (t == 0) lstar[b] = ix & (LEN - 1);
}

__device__ __forceinline__ float block_sum_128(float v, float* red, int t) {
  red[t] = v; __syncthreads();
  #pragma unroll
  for (int st = 64; st > 0; st >>= 1) { if (t < st) red[t] += red[t + st]; __syncthreads(); }
  float r = red[0]; __syncthreads();
  return r;
}

// common front-end of _snr on column l*: col, rd_col, idx, peak, noise
__device__ __forceinline__ void snr_column(const _Float16* xmr, const _Float16* xmi,
                                           const _Float16* Fr, const _Float16* Fi,
                                           const float* cre, const float* cim,
                                           int b, int f, int ls,
                                           float* s_iqr, float* s_iqi, float* s_col,
                                           float* red, float* sh,
                                           float& rr, float& ri, float& xrv, float& xiv,
                                           int& idx, float& peak, float& noise) {
  int s = f;
  xrv = (float)xmr[((size_t)b * NPULSE + s) * LEN + ls];
  xiv = (float)xmi[((size_t)b * NPULSE + s) * LEN + ls];
  float cr = cre[b * NPULSE + s], ci = cim[b * NPULSE + s];
  s_iqr[s] = cr * xrv - ci * xiv;
  s_iqi[s] = cr * xiv + ci * xrv;
  __syncthreads();
  rr = 0.0f; ri = 0.0f;
  for (int ss = 0; ss < NPULSE; ss++) {
    float fr = (float)Fr[f * NPULSE + ss], fi = (float)Fi[f * NPULSE + ss];
    float ar = s_iqr[ss], ai = s_iqi[ss];
    rr = fmaf(fr, ar, rr); rr = fmaf(-fi, ai, rr);
    ri = fmaf(fr, ai, ri); ri = fmaf(fi, ar, ri);
  }
  float col = rr * rr + ri * ri;
  s_col[f] = col;
  float ang = 6.28318530717958647692f * (float)f / 128.0f;
  float sn, cs; __sincosf(ang, &sn, &cs);
  float zre = block_sum_128(col * cs, red, f);
  float zim = block_sum_128(col * sn, red, f);
  float csum = block_sum_128(col, red, f);
  if (f == 0) {
    float pk = atan2f(zim, zre) * (128.0f / 6.28318530717958647692f);
    int id = (int)rintf(pk);
    id = ((id % NPULSE) + NPULSE) % NPULSE;
    sh[0] = (float)id; sh[1] = csum;
  }
  __syncthreads();
  idx = (int)sh[0];
  peak = s_col[idx];
  noise = (sh[1] - peak) / (float)(NPULSE - 1);
}

// ---------------------------------------------------------------------------
// analytic gradient of sum_b snr wrt (w_mag, w_phase) + step update
__global__ __launch_bounds__(128)
void grad_update_kernel(const _Float16* __restrict__ xmr, const _Float16* __restrict__ xmi,
                        const _Float16* __restrict__ Fr, const _Float16* __restrict__ Fi,
                        const float* __restrict__ cre, const float* __restrict__ cim,
                        const float* __restrict__ pbuf, const int* __restrict__ lstar,
                        const float* __restrict__ lsm, const float* __restrict__ lsp,
                        float* __restrict__ wmag, float* __restrict__ wph, int step) {
  __shared__ float s_iqr[NPULSE], s_iqi[NPULSE], s_col[NPULSE];
  __shared__ float s_gr[NPULSE], s_gi[NPULSE];
  __shared__ float red[NPULSE];
  __shared__ float sh[2];
  int b = blockIdx.x, f = threadIdx.x, s = f;
  int ls = lstar[b];
  float rr, ri, xrv, xiv, peak, noise; int idx;
  snr_column(xmr, xmi, Fr, Fi, cre, cim, b, f, ls,
             s_iqr, s_iqi, s_col, red, sh, rr, ri, xrv, xiv, idx, peak, noise);

  const float C = 4.34294481903251827651f;     // 10/ln(10)
  float gcol = (f == idx) ? (C / peak) : (-C / ((float)(NPULSE - 1) * noise));
  s_gr[f] = 2.0f * rr * gcol;                  // grad_rd (real-pair convention)
  s_gi[f] = 2.0f * ri * gcol;
  __syncthreads();

  // grad_iq[s] = sum_f conj(F[f,s]) * grad_rd[f]
  float gir = 0.0f, gii = 0.0f;
  for (int ff = 0; ff < NPULSE; ff++) {
    float fr = (float)Fr[ff * NPULSE + s], fi = (float)Fi[ff * NPULSE + s];
    gir = fmaf(fr, s_gr[ff], gir); gir = fmaf(fi, s_gi[ff], gir);
    gii = fmaf(fr, s_gi[ff], gii); gii = fmaf(-fi, s_gr[ff], gii);
  }
  // grad_c = conj(x) * grad_iq
  float gcr = xrv * gir + xiv * gii;
  float gci = xrv * gii - xiv * gir;
  // dc/dm = e^{-i th} = c / nm ; dc/dth = -i*c
  float p = pbuf[b * NPULSE + s];
  float nm = (float)NPULSE * p;
  float cr = cre[b * NPULSE + s], ci = cim[b * NPULSE + s];
  float gm = (gcr * cr + gci * ci) / nm;       // Re(conj(g)*e^{-ith})
  float gth = ci * gcr - cr * gci;             // Im(c*conj(g))
  // softmax backward: g_w = S*(p*gm - p*sum(p*gm))
  float dot = block_sum_128(p * gm, red, f);
  float gwm = (float)NPULSE * (p * gm - p * dot);
  float wpv = wph[b * NPULSE + s];
  float gwp = gth / (1.0f + wpv * wpv);
  float lrm = __expf(lsm[s * NSTEP + step] * 2.30258509299404568402f);   // 10^x
  float lrp = __expf(lsp[s * NSTEP + step] * 2.30258509299404568402f);
  wmag[b * NPULSE + s] += lrm * gwm;
  wph[b * NPULSE + s] += lrp * gwp;
}

// ---------------------------------------------------------------------------
__global__ __launch_bounds__(128)
void snr_kernel(const _Float16* __restrict__ xmr, const _Float16* __restrict__ xmi,
                const _Float16* __restrict__ Fr, const _Float16* __restrict__ Fi,
                const float* __restrict__ cre, const float* __restrict__ cim,
                const int* __restrict__ lstar, float* __restrict__ snr) {
  __shared__ float s_iqr[NPULSE], s_iqi[NPULSE], s_col[NPULSE];
  __shared__ float red[NPULSE];
  __shared__ float sh[2];
  int b = blockIdx.x, f = threadIdx.x;
  int ls = lstar[b];
  float rr, ri, xrv, xiv, peak, noise; int idx;
  snr_column(xmr, xmi, Fr, Fi, cre, cim, b, f, ls,
             s_iqr, s_iqi, s_col, red, sh, rr, ri, xrv, xiv, idx, peak, noise);
  if (f == 0) snr[b] = 10.0f * log10f(peak) - 10.0f * log10f(noise);
}

__global__ void mean_kernel(const float* __restrict__ snr, float* __restrict__ out) {
  int t = threadIdx.x;                         // 32 threads
  float v = snr[t];
  #pragma unroll
  for (int off = 16; off > 0; off >>= 1) v += __shfl_down(v, off);
  if (t == 0) out[0] = v / (float)BATCH;
}

// ---------------------------------------------------------------------------
extern "C" void kernel_launch(void* const* d_in, const int* in_sizes, int n_in,
                              void* d_out, int out_size, void* d_ws, size_t ws_size,
                              hipStream_t stream) {
  const float* x_re = (const float*)d_in[0];
  const float* x_im = (const float*)d_in[1];
  const float* k_re = (const float*)d_in[2];
  const float* k_im = (const float*)d_in[3];
  const float* lsm  = (const float*)d_in[4];
  const float* lsp  = (const float*)d_in[5];
  const float* wp0  = (const float*)d_in[6];
  float* out = (float*)d_out;

  char* ws = (char*)d_ws;
  size_t o = 0;
  const size_t NX = (size_t)BATCH * NPULSE * LEN;
  _Float16* xmr = (_Float16*)(ws + o); o += NX * sizeof(_Float16);
  _Float16* xmi = (_Float16*)(ws + o); o += NX * sizeof(_Float16);
  _Float16* Fr  = (_Float16*)(ws + o); o += (size_t)NPULSE * NPULSE * sizeof(_Float16);
  _Float16* Fi  = (_Float16*)(ws + o); o += (size_t)NPULSE * NPULSE * sizeof(_Float16);
  float* cre  = (float*)(ws + o); o += (size_t)BATCH * NPULSE * sizeof(float);
  float* cim  = (float*)(ws + o); o += (size_t)BATCH * NPULSE * sizeof(float);
  float* pbuf = (float*)(ws + o); o += (size_t)BATCH * NPULSE * sizeof(float);
  float* wmag = (float*)(ws + o); o += (size_t)BATCH * NPULSE * sizeof(float);
  float* wph  = (float*)(ws + o); o += (size_t)BATCH * NPULSE * sizeof(float);
  float* pmv  = (float*)(ws + o); o += (size_t)BATCH * NTILES * sizeof(float);
  int*   pmi  = (int*)(ws + o);   o += (size_t)BATCH * NTILES * sizeof(int);
  int*   lst  = (int*)(ws + o);   o += (size_t)BATCH * sizeof(int);
  float* snrb = (float*)(ws + o); o += (size_t)BATCH * sizeof(float);

  init_w_kernel<<<(BATCH * NPULSE + 255) / 256, 256, 0, stream>>>(wp0, wmag, wph);
  dft_init_kernel<<<(NPULSE * NPULSE + 255) / 256, 256, 0, stream>>>(Fr, Fi);
  matched_filter_kernel<<<BATCH * NPULSE, 256, 0, stream>>>(x_re, x_im, k_re, k_im, xmr, xmi);

  for (int step = 0; step < NSTEP; step++) {
    compute_c_kernel<<<BATCH, NPULSE, 0, stream>>>(wmag, wph, cre, cim, pbuf);
    dft_energy_kernel<<<BATCH * NTILES, 256, 0, stream>>>(xmr, xmi, Fr, Fi, cre, cim, pmv, pmi);
    reduce_lstar_kernel<<<BATCH, 32, 0, stream>>>(pmv, pmi, lst);
    grad_update_kernel<<<BATCH, NPULSE, 0, stream>>>(xmr, xmi, Fr, Fi, cre, cim, pbuf, lst,
                                                     lsm, lsp, wmag, wph, step);
  }
  compute_c_kernel<<<BATCH, NPULSE, 0, stream>>>(wmag, wph, cre, cim, pbuf);
  dft_energy_kernel<<<BATCH * NTILES, 256, 0, stream>>>(xmr, xmi, Fr, Fi, cre, cim, pmv, pmi);
  reduce_lstar_kernel<<<BATCH, 32, 0, stream>>>(pmv, pmi, lst);
  snr_kernel<<<BATCH, NPULSE, 0, stream>>>(xmr, xmi, Fr, Fi, cre, cim, lst, snrb);
  mean_kernel<<<1, 32, 0, stream>>>(snrb, out);
}
